// GraphPropagation_4260607558104
// MI455X (gfx1250) — matrix operand order repaired and checked by reading, not compiled, not run on hardware
//
#include <hip/hip_runtime.h>

// Problem constants (from reference): B=2, N=2048, Q=256, 15 iterations.
#define B_ 2
#define N_ 2048
#define Q_ 256
#define ITERS_ 15

// Padded LDS layout of the staged A strip (16 rows x 2048 f16):
// TDM inserts a 16B pad after every 1024B -> chunk pitch 1040B, row pitch
// 4160B (row bank-skew of 16 banks; chunks stay 16B aligned for ds_load_b128).
#define CHUNK_PITCH 1040
#define ROW_PITCH   4160
#define A_STRIP_LDS (16 * ROW_PITCH)   // 66560 bytes
#define RED_FLOATS  (128 + 128 + 16 + 16 + 16)

typedef _Float16 half_t;
typedef __attribute__((ext_vector_type(16))) _Float16 v16h;
typedef __attribute__((ext_vector_type(8)))  _Float16 v8h;
typedef __attribute__((ext_vector_type(8)))  float    v8f;
typedef __attribute__((ext_vector_type(4)))  unsigned int u32x4;
typedef __attribute__((ext_vector_type(8)))  unsigned int u32x8;

__device__ __forceinline__ v16h cat16(v8h lo, v8h hi) {
  return __builtin_shufflevector(lo, hi, 0,1,2,3,4,5,6,7,8,9,10,11,12,13,14,15);
}

// ---------------------------------------------------------------------------
// TDM: stage one contiguous 64KB A strip (16 rows x 2048 f16) into LDS@0 with
// 16B padding per 1024B chunk. 1-D tile, D# groups 0/1 only (ISA 8.3/8.4).
// Issued once per wave-0; EXEC is ignored by TENSOR ops.
// ---------------------------------------------------------------------------
__device__ __forceinline__ void tdm_stage_strip(const half_t* gsrc) {
  const unsigned long long ga = (unsigned long long)gsrc;
  u32x4 g0;
  g0[0] = 1u;                                   // count=1 (valid), user mode
  g0[1] = 0u;                                   // lds_addr = 0 (dynamic LDS base)
  g0[2] = (unsigned int)ga;                     // global_addr[31:0]
  g0[3] = (unsigned int)((ga >> 32) & 0x1ffffffu) | (2u << 30); // addr[56:32]|type=2
  u32x8 g1;
  // data_size=1 (2B) | pad_enable | pad_interval=7 (256 DW) | pad_amount=3 (4 DW)
  g1[0] = 0x00010000u | (1u << 20) | (7u << 22) | (3u << 25);
  g1[1] = 0x80000000u;                          // tensor_dim0[15:0]=32768 @bits63:48
  g1[2] = 0x00010000u;                          // tensor_dim0[31:16]=0, tensor_dim1=1
  g1[3] = 0x80000000u;                          // tile_dim0=32768 @bits127:112
  g1[4] = 1u;                                   // tile_dim1=1, tile_dim2=0
  g1[5] = 32768u;                               // tensor_dim0_stride[31:0]
  g1[6] = 0u;
  g1[7] = 0u;
  asm volatile("tensor_load_to_lds %0, %1" :: "s"(g0), "s"(g1) : "memory");
}

// ---------------------------------------------------------------------------
// 1) Threshold + transpose adj -> AeT[b][m][n], AiT[b][m][n] in f16.
// ---------------------------------------------------------------------------
__global__ void k_prep_adj(const float* __restrict__ adj,
                           half_t* __restrict__ AeT, half_t* __restrict__ AiT) {
  __shared__ half_t se[32][33];
  __shared__ half_t si[32][33];
  const int b  = blockIdx.z;
  const int n0 = blockIdx.y * 32;
  const int m0 = blockIdx.x * 32;
  const int tx = threadIdx.x, ty = threadIdx.y; // 32 x 8
  const float* src = adj + ((size_t)b * N_ + n0) * N_ + m0;
#pragma unroll
  for (int k = 0; k < 4; ++k) {
    const int i = ty + 8 * k;
    const float a = src[(size_t)i * N_ + tx];
    const float e  = (a > 0.5f) ? a : 0.0f;
    const float v  = 1.0f - a;
    const float ii = (v > 0.5f) ? v : 0.0f;
    se[i][tx] = (half_t)e;
    si[i][tx] = (half_t)ii;
  }
  __syncthreads();
  half_t* oe = AeT + ((size_t)b * N_ + m0) * N_ + n0;
  half_t* oi = AiT + ((size_t)b * N_ + m0) * N_ + n0;
#pragma unroll
  for (int k = 0; k < 4; ++k) {
    const int i = ty + 8 * k;
    oe[(size_t)i * N_ + tx] = se[tx][i];
    oi[(size_t)i * N_ + tx] = si[tx][i];
  }
}

// ---------------------------------------------------------------------------
// 2) inv[b][m] = 1 / max(sum_n AT[b][m][n], 1)  (deterministic tree reduce)
// ---------------------------------------------------------------------------
__global__ void k_colsum_inv(const half_t* __restrict__ AT, float* __restrict__ inv) {
  const int m = blockIdx.x, b = blockIdx.y;
  const half_t* row = AT + ((size_t)b * N_ + m) * N_;
  const int tid = threadIdx.x;
  float s = 0.0f;
#pragma unroll
  for (int k = tid; k < N_; k += 256) s += (float)row[k];
  __shared__ float red[256];
  red[tid] = s;
  __syncthreads();
  for (int off = 128; off > 0; off >>= 1) {
    if (tid < off) red[tid] += red[tid + off];
    __syncthreads();
  }
  if (tid == 0) inv[(size_t)b * N_ + m] = 1.0f / fmaxf(red[0], 1.0f);
}

// ---------------------------------------------------------------------------
// 3) h state = h0 ; Ht[b][q][n] = f16( h0[b][n][q] * act0[b][n] )
// ---------------------------------------------------------------------------
__global__ void k_init_state(const float* __restrict__ h0,
                             const float* __restrict__ act0,
                             float* __restrict__ h, half_t* __restrict__ Ht) {
  const size_t idx = (size_t)blockIdx.x * blockDim.x + threadIdx.x;
  if (idx >= (size_t)B_ * N_ * Q_) return;
  const int q = (int)(idx % Q_);
  const size_t t = idx / Q_;
  const int n = (int)(t % N_);
  const int b = (int)(t / N_);
  const float v = h0[idx];
  h[idx] = v;
  Ht[((size_t)b * Q_ + q) * N_ + n] = (half_t)(v * act0[(size_t)b * N_ + n]);
}

// ---------------------------------------------------------------------------
// GEMM core, A from padded LDS strip, B from global f16 Ht (L2-resident):
//   acc[m][q] = sum_n A[m][n] * Ht[q][n],  K = 2048 in 4 chunks x 16 steps.
// A frag (16x32): lane<16 row m holds K 0..15, lane>=16 K 16..31 -> two 16B
// ds loads at +2s+32hi / +2s+16+32hi inside a 1024B chunk (never crosses pad).
// B frag (32x16): lane = q col (lane&15), 16 contiguous halfs at kb=(lane>>4)*16.
// ---------------------------------------------------------------------------
__device__ __forceinline__ void gemm_tiles_lds(const char* sA, const half_t* __restrict__ HT,
                                               int qw, int lane, v8f& acc0, v8f& acc1) {
  const int row = lane & 15, hi = lane >> 4;
  const char* aRow = sA + row * ROW_PITCH + hi * 32;
  const half_t* b0r = HT + (size_t)(qw + row) * N_ + hi * 16;
  const half_t* b1r = HT + (size_t)(qw + 16 + row) * N_ + hi * 16;
#pragma unroll
  for (int c = 0; c < 4; ++c) {                 // 512-half padded chunks
    const char* aC = aRow + c * CHUNK_PITCH;
    const half_t* b0c = b0r + c * 512;
    const half_t* b1c = b1r + c * 512;
#pragma unroll 4
    for (int s = 0; s < 512; s += 32) {
      v16h a  = cat16(*(const v8h*)(aC + 2 * s), *(const v8h*)(aC + 2 * s + 16));
      v16h b0 = cat16(*(const v8h*)(b0c + s),   *(const v8h*)(b0c + s + 8));
      v16h b1 = cat16(*(const v8h*)(b1c + s),   *(const v8h*)(b1c + s + 8));
      acc0 = __builtin_amdgcn_wmma_f32_16x16x32_f16(false, a, false, b0, (short)0,
                                                    acc0, false, false);
      acc1 = __builtin_amdgcn_wmma_f32_16x16x32_f16(false, a, false, b1, (short)0,
                                                    acc1, false, false);
    }
  }
}

// ---------------------------------------------------------------------------
// 4) Excite: h += (Ae^T h_act) * neInv ; Ht_out = f16(h * act).
// ---------------------------------------------------------------------------
__global__ void __launch_bounds__(256)
k_excite(const half_t* __restrict__ AeT, const half_t* __restrict__ HtIn,
         const float* __restrict__ act0,
         float* __restrict__ h, const float* __restrict__ neInv,
         int ne_is_one, int act_onehot, half_t* __restrict__ HtOut) {
  extern __shared__ char smem[];
  const int m0 = blockIdx.x * 16, b = blockIdx.y;
  const int tid = threadIdx.x, wave = tid >> 5, lane = tid & 31;
  const int qw = wave * 32;
  const half_t* AT = AeT + ((size_t)b * N_ + m0) * N_;   // contiguous 64KB strip
  const half_t* HT = HtIn + (size_t)b * Q_ * N_;

  if (wave == 0) {                 // TDM copy + per-wave tensor count wait
    tdm_stage_strip(AT);
    __builtin_amdgcn_s_wait_tensorcnt(0);
  }
  __syncthreads();

  v8f acc0 = {}, acc1 = {};
  gemm_tiles_lds(smem, HT, qw, lane, acc0, acc1);

  const int row = lane & 15, hi = lane >> 4;
#pragma unroll
  for (int r = 0; r < 8; ++r) {
    const int m = m0 + r + 8 * hi;
    const float inv = ne_is_one ? 1.0f : neInv[(size_t)b * N_ + m];
    const float sc  = act_onehot ? act0[(size_t)b * N_ + m] : 1.0f;
    {
      const int q = qw + row;
      const size_t idx = ((size_t)b * N_ + m) * Q_ + q;
      const float hn = h[idx] + acc0[r] * inv;
      h[idx] = hn;
      HtOut[((size_t)b * Q_ + q) * N_ + m] = (half_t)(hn * sc);
    }
    {
      const int q = qw + 16 + row;
      const size_t idx = ((size_t)b * N_ + m) * Q_ + q;
      const float hn = h[idx] + acc1[r] * inv;
      h[idx] = hn;
      HtOut[((size_t)b * Q_ + q) * N_ + m] = (half_t)(hn * sc);
    }
  }
}

// ---------------------------------------------------------------------------
// 5) Inhibit: i_eff = (Ai^T h_act) * niInv ; per-row projection, relu,
//    L2-normalize over Q ; emit out[it], h, Ht(next). Deterministic
//    butterfly (within 16-lane halves = same m) + LDS cross-wave reduce.
// ---------------------------------------------------------------------------
__global__ void __launch_bounds__(256)
k_inhibit(const half_t* __restrict__ AiT, const half_t* __restrict__ HtIn,
          float* __restrict__ h, const float* __restrict__ niInv, int ni_is_one,
          float* __restrict__ out, half_t* __restrict__ HtOut) {
  extern __shared__ char smem[];
  float* s_pd  = (float*)(smem + A_STRIP_LDS);   // [8][16]
  float* s_pu  = s_pd + 128;                     // [8][16]
  float* s_dot = s_pu + 128;                     // [16]
  float* s_un  = s_dot + 16;                     // [16]
  float* s_nrm = s_un + 16;                      // [16]

  const int m0 = blockIdx.x * 16, b = blockIdx.y;
  const int tid = threadIdx.x, wave = tid >> 5, lane = tid & 31;
  const int qw = wave * 32;
  const half_t* AT = AiT + ((size_t)b * N_ + m0) * N_;
  const half_t* HT = HtIn + (size_t)b * Q_ * N_;

  if (wave == 0) {
    tdm_stage_strip(AT);
    __builtin_amdgcn_s_wait_tensorcnt(0);
  }
  __syncthreads();

  v8f acc0 = {}, acc1 = {};
  gemm_tiles_lds(smem, HT, qw, lane, acc0, acc1);

  const int row = lane & 15, hi = lane >> 4;
  float ie0[8], ie1[8], hv0[8], hv1[8];
#pragma unroll
  for (int r = 0; r < 8; ++r) {
    const int m = m0 + r + 8 * hi;
    const float inv = ni_is_one ? 1.0f : niInv[(size_t)b * N_ + m];
    ie0[r] = acc0[r] * inv;
    ie1[r] = acc1[r] * inv;
    hv0[r] = h[((size_t)b * N_ + m) * Q_ + qw + row];
    hv1[r] = h[((size_t)b * N_ + m) * Q_ + qw + 16 + row];
  }

  // dot = sum_q h*i ; unorm = sum_q i*i (per m row)
#pragma unroll
  for (int r = 0; r < 8; ++r) {
    float pd = hv0[r] * ie0[r] + hv1[r] * ie1[r];
    float pu = ie0[r] * ie0[r] + ie1[r] * ie1[r];
#pragma unroll
    for (int msk = 1; msk < 16; msk <<= 1) {     // stays in each 16-lane half
      pd += __shfl_xor(pd, msk);
      pu += __shfl_xor(pu, msk);
    }
    if (row == 0) { s_pd[wave * 16 + r + 8 * hi] = pd; s_pu[wave * 16 + r + 8 * hi] = pu; }
  }
  __syncthreads();
  if (tid < 16) {
    float d = 0.0f, u = 0.0f;
#pragma unroll
    for (int w = 0; w < 8; ++w) { d += s_pd[w * 16 + tid]; u += s_pu[w * 16 + tid]; }
    s_dot[tid] = d;
    s_un[tid]  = u;
  }
  __syncthreads();

  float rl0[8], rl1[8];
#pragma unroll
  for (int r = 0; r < 8; ++r) {
    const int mr = r + 8 * hi;
    const float sc = s_dot[mr] / (s_un[mr] + 1e-12f);
    rl0[r] = fmaxf(hv0[r] - sc * ie0[r], 0.0f);
    rl1[r] = fmaxf(hv1[r] - sc * ie1[r], 0.0f);
  }

  // L2 norm over q (s_pd reusable: phase-1 readers done before 2nd barrier)
#pragma unroll
  for (int r = 0; r < 8; ++r) {
    float pn = rl0[r] * rl0[r] + rl1[r] * rl1[r];
#pragma unroll
    for (int msk = 1; msk < 16; msk <<= 1) pn += __shfl_xor(pn, msk);
    if (row == 0) s_pd[wave * 16 + r + 8 * hi] = pn;
  }
  __syncthreads();
  if (tid < 16) {
    float s = 0.0f;
#pragma unroll
    for (int w = 0; w < 8; ++w) s += s_pd[w * 16 + tid];
    s_nrm[tid] = fmaxf(sqrtf(s), 1e-8f);
  }
  __syncthreads();

#pragma unroll
  for (int r = 0; r < 8; ++r) {
    const int m = m0 + r + 8 * hi;
    const float dn = s_nrm[r + 8 * hi];
    {
      const int q = qw + row;
      const float v = rl0[r] / dn;
      const size_t idx = ((size_t)b * N_ + m) * Q_ + q;
      out[idx] = v;
      h[idx] = v;
      HtOut[((size_t)b * Q_ + q) * N_ + m] = (half_t)v;   // next act == 1
    }
    {
      const int q = qw + 16 + row;
      const float v = rl1[r] / dn;
      const size_t idx = ((size_t)b * N_ + m) * Q_ + q;
      out[idx] = v;
      h[idx] = v;
      HtOut[((size_t)b * Q_ + q) * N_ + m] = (half_t)v;
    }
  }
}

// ---------------------------------------------------------------------------
extern "C" void kernel_launch(void* const* d_in, const int* in_sizes, int n_in,
                              void* d_out, int out_size, void* d_ws, size_t ws_size,
                              hipStream_t stream) {
  (void)in_sizes; (void)n_in; (void)out_size; (void)ws_size;
  const float* h0   = (const float*)d_in[0];   // [B,N,Q] f32
  const float* adj  = (const float*)d_in[1];   // [B,N,N] f32
  const float* act0 = (const float*)d_in[2];   // [B,N,1] f32 (one-hot node 0)
  float* out = (float*)d_out;                  // [ITERS,B,N,Q] f32

  char* ws = (char*)d_ws;
  size_t off = 0;
  auto carve = [&](size_t bytes) -> void* {
    void* p = (void*)(ws + off);
    off = (off + bytes + 255) & ~(size_t)255;
    return p;
  };
  half_t* AeT  = (half_t*)carve((size_t)B_ * N_ * N_ * sizeof(half_t)); // 16.8 MB
  half_t* AiT  = (half_t*)carve((size_t)B_ * N_ * N_ * sizeof(half_t)); // 16.8 MB
  half_t* HtA  = (half_t*)carve((size_t)B_ * Q_ * N_ * sizeof(half_t)); // 2 MB
  half_t* HtB  = (half_t*)carve((size_t)B_ * Q_ * N_ * sizeof(half_t)); // 2 MB
  float*  hbuf = (float*) carve((size_t)B_ * N_ * Q_ * sizeof(float));  // 4 MB
  float*  neInv = (float*)carve((size_t)B_ * N_ * sizeof(float));
  float*  niInv = (float*)carve((size_t)B_ * N_ * sizeof(float));

  k_prep_adj<<<dim3(N_ / 32, N_ / 32, B_), dim3(32, 8), 0, stream>>>(adj, AeT, AiT);
  k_colsum_inv<<<dim3(N_, B_), 256, 0, stream>>>(AeT, neInv);
  k_colsum_inv<<<dim3(N_, B_), 256, 0, stream>>>(AiT, niInv);
  k_init_state<<<(B_ * N_ * Q_) / 256, 256, 0, stream>>>(h0, act0, hbuf, HtA);

  const unsigned ldsE = A_STRIP_LDS;                       // 66560 B
  const unsigned ldsI = A_STRIP_LDS + RED_FLOATS * 4;      // + reduction scratch

  for (int it = 0; it < ITERS_; ++it) {
    const int first = (it == 0) ? 1 : 0;   // n_e = n_i = 1 and act one-hot
    k_excite<<<dim3(N_ / 16, B_), 256, ldsE, stream>>>(
        AeT, HtA, act0, hbuf, neInv, first, first, HtB);
    k_inhibit<<<dim3(N_ / 16, B_), 256, ldsI, stream>>>(
        AiT, HtB, hbuf, niInv, first,
        out + (size_t)it * B_ * N_ * Q_, HtA);
  }
}
